// Encoder_8821862826192
// MI455X (gfx1250) — matrix operand order repaired and checked by reading, not compiled, not run on hardware
//
#include <hip/hip_runtime.h>
#include <hip/hip_bf16.h>

// Persistent fused 2-layer GRU encoder for gfx1250 (MI455X).
// One wave == one 16-row batch tile for all 1024 timesteps.
// Per-step GEMMs via v_wmma_f32_16x16x32_f16 with bias/input folded into
// augmented K columns. No inter-wave synchronization in the time loop.

typedef _Float16 v16h __attribute__((ext_vector_type(16)));
typedef _Float16 v8h  __attribute__((ext_vector_type(8)));
typedef float    v8f  __attribute__((ext_vector_type(8)));

#define P1 72    // A1 row pitch in f16 (>=66 used cols, 16B-aligned rows: 144B)
#define P2 104   // A2 row pitch in f16 (>=98 used cols, 16B-aligned rows: 208B)
#define TT 1024
#define XCHUNK 32

__device__ __forceinline__ v8f vzero8() {
  v8f r;
#pragma unroll
  for (int i = 0; i < 8; ++i) r[i] = 0.0f;
  return r;
}

__device__ __forceinline__ v8f wmma_f16(v16h a, v16h b, v8f c) {
  // D = A(16x32 f16) * B(32x16 f16) + C(16x16 f32)
  return __builtin_amdgcn_wmma_f32_16x16x32_f16(
      /*neg_a=*/false, a, /*neg_b=*/false, b,
      /*c_mod=*/(short)0, c, /*reuse_a=*/false, /*reuse_b=*/false);
}

// A-matrix fragment gather from row-major f16 LDS panel.
// A layout (16-bit A 16x32): lane<16 -> M=lane, elems 0..7 = K kt*32+0..7,
// elems 8..15 = K kt*32+16..23; lane>=16 adds +8 to both K groups.
__device__ __forceinline__ v16h load_afrag(const _Float16* base, int pitch,
                                           int mt, int kt, int li, int hi) {
  const int m  = mt * 16 + li;
  const int o1 = m * pitch + kt * 32 + hi * 8;  // 8 f16 = 16B, 16B-aligned
  v8h lo = *(const v8h*)(base + o1);
  v8h hg = *(const v8h*)(base + o1 + 16);
  v16h r;
#pragma unroll
  for (int i = 0; i < 8; ++i) { r[i] = lo[i]; r[i + 8] = hg[i]; }
  return r;
}

// B-matrix fragment: lane%16 = batch column N; elems 0..7 = K rows rowbase+8h+0..7,
// elems 8..15 = K rows rowbase+16+8h+0..7 (h = lane/16), matching the A layout.
__device__ __forceinline__ v16h load_bfrag(const _Float16* bb, int rowbase, int lane) {
  const int n = lane & 15, h = lane >> 4;
  v16h r;
#pragma unroll
  for (int s = 0; s < 8; ++s) {
    r[s]     = bb[(rowbase + 8 * h + s) * 16 + n];
    r[s + 8] = bb[(rowbase + 16 + 8 * h + s) * 16 + n];
  }
  return r;
}

__device__ __forceinline__ float sigm(float x) {
  return __builtin_amdgcn_rcpf(1.0f + __expf(-x));
}
__device__ __forceinline__ float tanh_fast(float x) {
  float e = __expf(-2.0f * x);
  return (1.0f - e) * __builtin_amdgcn_rcpf(1.0f + e);
}

__global__ __launch_bounds__(64, 1) void gru2_encoder_kernel(
    const float* __restrict__ x,
    const float* __restrict__ Wih1, const float* __restrict__ Whh1,
    const float* __restrict__ bih1, const float* __restrict__ bhh1,
    const float* __restrict__ Wih2, const float* __restrict__ Whh2,
    const float* __restrict__ bih2, const float* __restrict__ bhh2,
    float* __restrict__ out)
{
  // Augmented weight panel L1: 192 rows (r:0-63, z:64-127, n:128-191).
  //  cols 0-63 = W_hh1 ; col 64 = W_ih1 (x) ; col 65 = bias (r/z: bih+bhh,
  //  n: bih only -> goes to gx_n) ; cols 66.. = 0/pad.
  __shared__ __align__(32) _Float16 sA1[192 * P1];
  // Augmented weight panel L2: 96 rows (r:0-31, z:32-63, n:64-95).
  //  cols 0-31 = W_hh2 (h2 rows 0-31) ; cols 32-95 = W_ih2 (h1 rows 32-95) ;
  //  col 96 = 0 (kills x row) ; col 97 = bias (r/z: bih+bhh, n: bih) ; pad.
  __shared__ __align__(32) _Float16 sA2[96 * P2];
  __shared__ float sBh1n[64];   // b_hh1 for n gate (elementwise, inside r*(.))
  __shared__ float sBh2n[32];   // b_hh2 for n gate
  // Unified per-wave B buffer, 128 rows x 16 batch cols (f16):
  //  rows 0-31 = h2 ; rows 32-95 = h1 ; row 96 = x_t ; row 97 = 1 ; 98-127 = 0
  __shared__ __align__(32) _Float16 sBbuf[2][128 * 16];
  __shared__ __align__(32) float    sXbuf[2][16 * XCHUNK];

  const int tid = threadIdx.x;

  // ---------------- one-time init: pack weights to f16 panels ----------------
  for (int i = tid; i < 192 * P1; i += 64) {
    int g = i / P1, c = i % P1;
    float v = 0.0f;
    if (c < 64)        v = Whh1[g * 64 + c];
    else if (c == 64)  v = Wih1[g];
    else if (c == 65)  v = (g < 128) ? (bih1[g] + bhh1[g]) : bih1[g];
    sA1[i] = (_Float16)v;
  }
  for (int i = tid; i < 96 * P2; i += 64) {
    int g = i / P2, c = i % P2;
    float v = 0.0f;
    if (c < 32)        v = Whh2[g * 32 + c];
    else if (c < 96)   v = Wih2[g * 64 + (c - 32)];
    else if (c == 97)  v = (g < 64) ? (bih2[g] + bhh2[g]) : bih2[g];
    sA2[i] = (_Float16)v;
  }
  if (tid < 64) sBh1n[tid] = bhh1[128 + tid];
  if (tid < 32) sBh2n[tid] = bhh2[64 + tid];
  for (int i = tid; i < 128 * 16; i += 64) {
    _Float16 v = (_Float16)0.0f;
    if (i >= 97 * 16 && i < 98 * 16) v = (_Float16)1.0f;  // ones row
    sBbuf[0][i] = v;
    sBbuf[1][i] = v;
  }
  __syncthreads();

  // ---------------- per-wave setup ----------------
  const int wave = tid >> 5;
  const int lane = tid & 31;
  const int li   = lane & 15;        // batch column within tile / A M-row
  const int hi   = lane >> 4;        // lane-half (adds +8 to D/A row index)
  const int batchBase = blockIdx.x * 32 + wave * 16;

  _Float16* bb = &sBbuf[wave][0];
  float*    xb = &sXbuf[wave][0];

  float h1o[4][8];                   // h1 (f32, D layout): [ht][j], row = ht*16+j+8*hi
  float h2o[2][8];                   // h2 (f32, D layout)
#pragma unroll
  for (int a = 0; a < 4; ++a)
#pragma unroll
    for (int j = 0; j < 8; ++j) h1o[a][j] = 0.0f;
#pragma unroll
  for (int a = 0; a < 2; ++a)
#pragma unroll
    for (int j = 0; j < 8; ++j) h2o[a][j] = 0.0f;

  const float* xrow = x + (size_t)(batchBase + li) * TT + hi * (XCHUNK / 2);

  // ---------------- time loop (fully wave-local, no barriers) ----------------
  for (int t = 0; t < TT; ++t) {
    if ((t & (XCHUNK - 1)) == 0) {
      // stage x[batch tile, t..t+31] into LDS; each lane moves 16 floats
      const float* xs = xrow + t;
      float* xd = xb + li * XCHUNK + hi * (XCHUNK / 2);
#pragma unroll
      for (int i = 0; i < 4; ++i)
        *(float4*)(xd + i * 4) = *(const float4*)(xs + i * 4);
      if (t + XCHUNK < TT) __builtin_prefetch(xs + XCHUNK, 0, 0);
    }
    // x_t for this lane's batch column -> B row 96 (f16)
    float xv = xb[li * XCHUNK + (t & (XCHUNK - 1))];
    bb[96 * 16 + li] = (_Float16)xv;

    v16h fH1lo = load_bfrag(bb, 32, lane);   // h1 rows 0..31
    v16h fH1hi = load_bfrag(bb, 64, lane);   // h1 rows 32..63
    v16h fAug  = load_bfrag(bb, 96, lane);   // [x, 1, 0...]

    // ================= Layer 1: gates^T = A1 * [h1; x; 1] =================
#pragma unroll
    for (int ht = 0; ht < 4; ++ht) {
      v8f aR = vzero8(), aZ = vzero8(), aGH = vzero8(), aGX = vzero8();
      aR  = wmma_f16(load_afrag(sA1, P1, ht,     0, li, hi), fH1lo, aR);
      aR  = wmma_f16(load_afrag(sA1, P1, ht,     1, li, hi), fH1hi, aR);
      aR  = wmma_f16(load_afrag(sA1, P1, ht,     2, li, hi), fAug,  aR);
      aZ  = wmma_f16(load_afrag(sA1, P1, ht + 4, 0, li, hi), fH1lo, aZ);
      aZ  = wmma_f16(load_afrag(sA1, P1, ht + 4, 1, li, hi), fH1hi, aZ);
      aZ  = wmma_f16(load_afrag(sA1, P1, ht + 4, 2, li, hi), fAug,  aZ);
      aGH = wmma_f16(load_afrag(sA1, P1, ht + 8, 0, li, hi), fH1lo, aGH);
      aGH = wmma_f16(load_afrag(sA1, P1, ht + 8, 1, li, hi), fH1hi, aGH);
      aGX = wmma_f16(load_afrag(sA1, P1, ht + 8, 2, li, hi), fAug,  aGX);
#pragma unroll
      for (int j = 0; j < 8; ++j) {
        int m = ht * 16 + j + hi * 8;                 // hidden row 0..63
        float r = sigm(aR[j]);
        float z = sigm(aZ[j]);
        float n = tanh_fast(aGX[j] + r * (aGH[j] + sBh1n[m]));
        float h = (1.0f - z) * n + z * h1o[ht][j];
        h1o[ht][j] = h;
        bb[(32 + m) * 16 + li] = (_Float16)h;         // h1 rows 32..95
      }
    }

    // ================= Layer 2: gates^T = A2 * [h2; h1_new; x; 1] =========
    v16h fH2 = load_bfrag(bb, 0, lane);      // h2 rows 0..31
    fH1lo = load_bfrag(bb, 32, lane);        // fresh h1 (just stored, DS in-order)
    fH1hi = load_bfrag(bb, 64, lane);
#pragma unroll
    for (int ht = 0; ht < 2; ++ht) {
      v8f aR = vzero8(), aZ = vzero8(), aGH = vzero8(), aGX = vzero8();
      aR  = wmma_f16(load_afrag(sA2, P2, ht,     0, li, hi), fH2,   aR);
      aR  = wmma_f16(load_afrag(sA2, P2, ht,     1, li, hi), fH1lo, aR);
      aR  = wmma_f16(load_afrag(sA2, P2, ht,     2, li, hi), fH1hi, aR);
      aR  = wmma_f16(load_afrag(sA2, P2, ht,     3, li, hi), fAug,  aR);
      aZ  = wmma_f16(load_afrag(sA2, P2, ht + 2, 0, li, hi), fH2,   aZ);
      aZ  = wmma_f16(load_afrag(sA2, P2, ht + 2, 1, li, hi), fH1lo, aZ);
      aZ  = wmma_f16(load_afrag(sA2, P2, ht + 2, 2, li, hi), fH1hi, aZ);
      aZ  = wmma_f16(load_afrag(sA2, P2, ht + 2, 3, li, hi), fAug,  aZ);
      aGH = wmma_f16(load_afrag(sA2, P2, ht + 4, 0, li, hi), fH2,   aGH);
      aGX = wmma_f16(load_afrag(sA2, P2, ht + 4, 1, li, hi), fH1lo, aGX);
      aGX = wmma_f16(load_afrag(sA2, P2, ht + 4, 2, li, hi), fH1hi, aGX);
      aGX = wmma_f16(load_afrag(sA2, P2, ht + 4, 3, li, hi), fAug,  aGX);
#pragma unroll
      for (int j = 0; j < 8; ++j) {
        int m = ht * 16 + j + hi * 8;                 // hidden row 0..31
        float r = sigm(aR[j]);
        float z = sigm(aZ[j]);
        float n = tanh_fast(aGX[j] + r * (aGH[j] + sBh2n[m]));
        float h = (1.0f - z) * n + z * h2o[ht][j];
        h2o[ht][j] = h;
        bb[m * 16 + li] = (_Float16)h;                // h2 rows 0..31
      }
    }
  }

  // ---------------- write final h2 [B, 32] ----------------
#pragma unroll
  for (int ht = 0; ht < 2; ++ht)
#pragma unroll
    for (int j = 0; j < 8; ++j) {
      int m = ht * 16 + j + hi * 8;
      out[(size_t)(batchBase + li) * 32 + m] = h2o[ht][j];
    }
}

extern "C" void kernel_launch(void* const* d_in, const int* in_sizes, int n_in,
                              void* d_out, int out_size, void* d_ws, size_t ws_size,
                              hipStream_t stream) {
  (void)in_sizes; (void)n_in; (void)out_size; (void)d_ws; (void)ws_size;
  const float* x    = (const float*)d_in[0];
  const float* Wih1 = (const float*)d_in[1];
  const float* Whh1 = (const float*)d_in[2];
  const float* bih1 = (const float*)d_in[3];
  const float* bhh1 = (const float*)d_in[4];
  const float* Wih2 = (const float*)d_in[5];
  const float* Whh2 = (const float*)d_in[6];
  const float* bih2 = (const float*)d_in[7];
  const float* bhh2 = (const float*)d_in[8];
  float* out = (float*)d_out;

  dim3 grid(32);    // 32 blocks x 2 waves x 16 batch rows = 1024
  dim3 block(64);
  gru2_encoder_kernel<<<grid, block, 0, stream>>>(
      x, Wih1, Whh1, bih1, bhh1, Wih2, Whh2, bih2, bhh2, out);
}
